// RowAttentionWithPairBias_41961830482531
// MI455X (gfx1250) — compile-verified, matrix operand order checked by reading, not run on hardware
//
#include <hip/hip_runtime.h>
#include <hip/hip_bf16.h>
#include <math.h>

typedef __attribute__((ext_vector_type(16))) _Float16 v16h;
typedef __attribute__((ext_vector_type(8)))  float    v8f;

#define S_  128
#define R_  384
#define CM  256
#define CZ  128
#define H_  8
#define D_  32

__device__ __forceinline__ int lane_id() { return threadIdx.x & 31; }

__device__ __forceinline__ v8f zero8() {
  v8f z;
#pragma unroll
  for (int i = 0; i < 8; ++i) z[i] = 0.f;
  return z;
}

__device__ __forceinline__ v8f wmma32(v16h a, v16h b, v8f c) {
  // D = A(16x32 f16) * B(32x16 f16) + C(16x16 f32)
  return __builtin_amdgcn_wmma_f32_16x16x32_f16(false, a, false, b, (short)0, c,
                                                false, false);
}

// A fragment: 16x32 f16 tile, row-major, leading dim ld.
// Lane L (0-15): row M=L, K = {0..7, 16..23}; lane 16-31: row M=L-16, K = {8..15, 24..31}.
// Also serves as B fragment when the B matrix is stored transposed (N x K row-major),
// since the B lane layout mirrors A with lane <-> N.
__device__ __forceinline__ v16h load_a_frag(const _Float16* base, int ld) {
  int lane = lane_id();
  int m = lane & 15, half = lane >> 4;
  const _Float16* rowp = base + m * ld + half * 8;
  v16h a;
#pragma unroll
  for (int e = 0; e < 16; ++e)
    a[e] = rowp[((e >> 3) << 4) + (e & 7)];
  return a;
}

__device__ __forceinline__ float wave_sum32(float x) {
#pragma unroll
  for (int off = 16; off > 0; off >>= 1) x += __shfl_xor(x, off, 32);
  return x;
}
__device__ __forceinline__ float half_sum16(float x) {
#pragma unroll
  for (int off = 8; off > 0; off >>= 1) x += __shfl_xor(x, off, 32);
  return x;
}
__device__ __forceinline__ float half_max16(float x) {
#pragma unroll
  for (int off = 8; off > 0; off >>= 1) x = fmaxf(x, __shfl_xor(x, off, 32));
  return x;
}

// ---------------------------------------------------------------- weights prep
__global__ void prep_weights_kernel(const float* wq, const float* wk, const float* wv,
                                    const float* wg, const float* wo,
                                    _Float16* wT, _Float16* woT) {
  int i = blockIdx.x * blockDim.x + threadIdx.x;
  const int NW = 4 * CM * CM;
  if (i < NW) {
    int proj = i >> 16;          // CM*CM == 65536
    int rem  = i & 65535;
    int hd   = rem >> 8;
    int c    = rem & 255;
    const float* w = (proj == 0) ? wq : (proj == 1) ? wk : (proj == 2) ? wv : wg;
    float scl = (proj == 0) ? 0.17677669529663689f : 1.0f;  // D^-0.5 folded into Wq
    wT[i] = (_Float16)(w[c * 256 + hd] * scl);              // wT[N=proj*256+hd][K=c]
  } else if (i < NW + CM * CM) {
    int j  = i - NW;
    int c  = j >> 8;
    int hd = j & 255;
    woT[j] = (_Float16)(wo[hd * 256 + c]);                  // woT[N=c][K=hd]
  }
}

// ---------------------------------------------------------------- pair bias z
// Stored TRANSPOSED: z[h, key_col, query_row] so attention reads are contiguous.
__global__ void pair_bias_kernel(const float* __restrict__ pair,
                                 const float* __restrict__ pn_g,
                                 const float* __restrict__ pn_b,
                                 const float* __restrict__ w_pair,
                                 float* __restrict__ z) {
  int wave = (blockIdx.x * blockDim.x + threadIdx.x) >> 5;
  int lane = lane_id();
  if (wave >= R_ * R_) return;
  const float* p = pair + (size_t)wave * CZ;
  float x[4], s = 0.f, ss = 0.f;
#pragma unroll
  for (int t = 0; t < 4; ++t) {
    float v = p[t * 32 + lane];
    x[t] = v; s += v; ss += v * v;
  }
  s = wave_sum32(s); ss = wave_sum32(ss);
  float mu = s * (1.f / CZ);
  float rsig = rsqrtf(ss * (1.f / CZ) - mu * mu + 1e-5f);
  float xn[4];
#pragma unroll
  for (int t = 0; t < 4; ++t) {
    int c = t * 32 + lane;
    xn[t] = (x[t] - mu) * rsig * pn_g[c] + pn_b[c];
  }
  int qi = wave / R_;            // query row i
  int kj = wave - qi * R_;       // key col j
  float zval = 0.f;
#pragma unroll
  for (int h = 0; h < H_; ++h) {
    float a = 0.f;
#pragma unroll
    for (int t = 0; t < 4; ++t) {
      int c = t * 32 + lane;
      a += xn[t] * w_pair[c * H_ + h];
    }
    a = wave_sum32(a);
    if (lane == h) zval = a;
  }
  if (lane < H_)
    z[(size_t)lane * R_ * R_ + (size_t)kj * R_ + qi] = zval;   // z[h, j, i]
}

// ---------------------------------------------------------------- MSA layernorm
__global__ void ln_msa_kernel(const float* __restrict__ msa,
                              const float* __restrict__ gg,
                              const float* __restrict__ bb,
                              _Float16* __restrict__ m) {
  int row = (blockIdx.x * blockDim.x + threadIdx.x) >> 5;
  int lane = lane_id();
  if (row >= S_ * R_) return;
  const float* p = msa + (size_t)row * CM;
  float x[8], s = 0.f, ss = 0.f;
#pragma unroll
  for (int t = 0; t < 8; ++t) {
    float v = p[t * 32 + lane];
    x[t] = v; s += v; ss += v * v;
  }
  s = wave_sum32(s); ss = wave_sum32(ss);
  float mu = s * (1.f / CM);
  float rsig = rsqrtf(ss * (1.f / CM) - mu * mu + 1e-5f);
  _Float16* o = m + (size_t)row * CM;
#pragma unroll
  for (int t = 0; t < 8; ++t) {
    int c = t * 32 + lane;
    o[c] = (_Float16)((x[t] - mu) * rsig * gg[c] + bb[c]);
  }
}

// ---------------------------------------------------------------- QKVG GEMM
// m[49152 x 256] x wT'[256 x 1024]
//   -> q,k [S,H,R,D] f16 (row-major tiles for A / K^T fragments)
//   -> vT,gT [S,H,D,R] f16 (transposed: contiguous B-fragment loads in attention)
__global__ void qkvg_gemm_kernel(const _Float16* __restrict__ m,
                                 const _Float16* __restrict__ wT,
                                 const float* __restrict__ bg,
                                 _Float16* __restrict__ q, _Float16* __restrict__ k,
                                 _Float16* __restrict__ vT, _Float16* __restrict__ gT) {
  int wave = (blockIdx.x * blockDim.x + threadIdx.x) >> 5;
  int lane = lane_id();
  int nt = wave & 63;
  int mt = wave >> 6;
  if (mt >= (S_ * R_) / 16) return;
  v8f acc = zero8();
  const _Float16* arow = m  + (size_t)(mt * 16) * CM;
  const _Float16* brow = wT + (size_t)(nt * 16) * CM;
#pragma unroll
  for (int k0 = 0; k0 < CM; k0 += 32)
    acc = wmma32(load_a_frag(arow + k0, CM), load_a_frag(brow + k0, CM), acc);

  int half = lane >> 4, n = lane & 15;
  int gn = nt * 16 + n;
  int proj = nt >> 4;           // uniform per wave
  int h = (gn >> 5) & 7;
  int d = gn & 31;
  // a 16-row M tile never crosses an s boundary (R % 16 == 0)
  int gm0 = mt * 16 + half * 8;
  int s = gm0 / R_;
  int r0 = gm0 - s * R_;
#pragma unroll
  for (int vi = 0; vi < 8; ++vi) {
    int r = r0 + vi;
    if (proj == 0) {
      q[(((size_t)s * H_ + h) * R_ + r) * D_ + d] = (_Float16)acc[vi];
    } else if (proj == 1) {
      k[(((size_t)s * H_ + h) * R_ + r) * D_ + d] = (_Float16)acc[vi];
    } else if (proj == 2) {
      vT[(((size_t)s * H_ + h) * D_ + d) * R_ + r] = (_Float16)acc[vi];  // contiguous over vi
    } else {
      float gv = 1.f / (1.f + __expf(-(acc[vi] + bg[h * D_ + d])));
      gT[(((size_t)s * H_ + h) * D_ + d) * R_ + r] = (_Float16)gv;       // contiguous over vi
    }
  }
}

// ---------------------------------------------------------------- attention
// One wave per (s, h, 16-row q tile). Flash-style online softmax over 32-key chunks.
__global__ void attn_kernel(const _Float16* __restrict__ q, const _Float16* __restrict__ k,
                            const _Float16* __restrict__ vT, const _Float16* __restrict__ gT,
                            const float* __restrict__ z, const float* __restrict__ mask,
                            _Float16* __restrict__ og) {
  __shared__ _Float16 plds[4][16][32];      // per-wave probs staging (A-layout refill)
  int w = threadIdx.x >> 5;
  int lane = lane_id();
  int t = blockIdx.x * 4 + w;
  const int NQ = R_ / 16;                   // 24
  int qt = t % NQ;
  int rem = t / NQ;
  int h = rem & 7;
  int s = rem >> 3;
  if (s >= S_) return;
  int q0 = qt * 16;

  const _Float16* Q  = q  + (((size_t)s * H_ + h) * R_) * D_;
  const _Float16* K  = k  + (((size_t)s * H_ + h) * R_) * D_;
  const _Float16* VT = vT + (((size_t)s * H_ + h) * D_) * R_;
  const _Float16* GT = gT + (((size_t)s * H_ + h) * D_) * R_;
  const float* Z = z + (size_t)h * R_ * R_;          // z[key, query] layout
  const float* M = mask + (size_t)s * R_;
  int half = lane >> 4, n = lane & 15;

  v16h aq = load_a_frag(Q + (size_t)q0 * D_, D_);
  v8f o0 = zero8(), o1 = zero8();
  float rmax[8], rsum[8];
#pragma unroll
  for (int i = 0; i < 8; ++i) { rmax[i] = -1e30f; rsum[i] = 0.f; }

  for (int kb = 0; kb < R_ / 32; ++kb) {
    int c0 = kb * 32;
    v16h bk0 = load_a_frag(K + (size_t)c0 * D_, D_);        // keys c0..c0+15
    v16h bk1 = load_a_frag(K + (size_t)(c0 + 16) * D_, D_); // keys c0+16..c0+31
    v8f s0 = wmma32(aq, bk0, zero8());
    v8f s1 = wmma32(aq, bk1, zero8());

    int col0 = c0 + n, col1 = c0 + 16 + n;
    float mb0 = 1e9f * (M[col0] - 1.f);
    float mb1 = 1e9f * (M[col1] - 1.f);
    // z[h, key, query]: 8 consecutive queries per lane -> coalesced vector loads
    const float* z0 = Z + (size_t)col0 * R_ + q0 + half * 8;
    const float* z1 = Z + (size_t)col1 * R_ + q0 + half * 8;
#pragma unroll
    for (int vi = 0; vi < 8; ++vi) {
      s0[vi] += z0[vi] + mb0;
      s1[vi] += z1[vi] + mb1;
    }
#pragma unroll
    for (int vi = 0; vi < 8; ++vi) {
      float mx = half_max16(fmaxf(s0[vi], s1[vi]));
      float nm = fmaxf(rmax[vi], mx);
      float sc = __expf(rmax[vi] - nm);
      float p0 = __expf(s0[vi] - nm);
      float p1 = __expf(s1[vi] - nm);
      rsum[vi] = rsum[vi] * sc + half_sum16(p0 + p1);
      rmax[vi] = nm;
      o0[vi] *= sc; o1[vi] *= sc;
      plds[w][half * 8 + vi][n]      = (_Float16)p0;
      plds[w][half * 8 + vi][16 + n] = (_Float16)p1;
    }
    // same-wave DS RAW: writes above feed the cross-lane A-fragment load below
    asm volatile("s_wait_dscnt 0" ::: "memory");
    v16h ap  = load_a_frag(&plds[w][0][0], 32);
    // V^T rows d0..d0+15 over keys c0..c0+31 == B fragment (lane <-> N=d)
    v16h bv0 = load_a_frag(VT + (size_t)0  * R_ + c0, R_);
    v16h bv1 = load_a_frag(VT + (size_t)16 * R_ + c0, R_);
    o0 = wmma32(ap, bv0, o0);
    o1 = wmma32(ap, bv1, o1);
  }

  // epilogue: normalize, gate, write og[row, h*32+d] (f16)
  const _Float16* g0p = GT + (size_t)n        * R_ + q0 + half * 8;  // contiguous
  const _Float16* g1p = GT + (size_t)(16 + n) * R_ + q0 + half * 8;  // contiguous
#pragma unroll
  for (int vi = 0; vi < 8; ++vi) {
    int row = q0 + half * 8 + vi;
    float inv = 1.f / rsum[vi];
    float g0 = (float)g0p[vi];
    float g1 = (float)g1p[vi];
    size_t dst = ((size_t)s * R_ + row) * 256 + h * D_;
    og[dst + n]      = (_Float16)(o0[vi] * inv * g0);
    og[dst + 16 + n] = (_Float16)(o1[vi] * inv * g1);
  }
}

// ---------------------------------------------------------------- output GEMM
__global__ void out_gemm_kernel(const _Float16* __restrict__ og,
                                const _Float16* __restrict__ woT,
                                const float* __restrict__ bo,
                                float* __restrict__ out) {
  int wave = (blockIdx.x * blockDim.x + threadIdx.x) >> 5;
  int lane = lane_id();
  int nt = wave & 15;
  int mt = wave >> 4;
  if (mt >= (S_ * R_) / 16) return;
  v8f acc = zero8();
  const _Float16* arow = og  + (size_t)(mt * 16) * 256;
  const _Float16* brow = woT + (size_t)(nt * 16) * 256;
#pragma unroll
  for (int k0 = 0; k0 < 256; k0 += 32)
    acc = wmma32(load_a_frag(arow + k0, 256), load_a_frag(brow + k0, 256), acc);
  int half = lane >> 4, n = lane & 15;
  float bias = bo[nt * 16 + n];
#pragma unroll
  for (int vi = 0; vi < 8; ++vi)
    out[(size_t)(mt * 16 + half * 8 + vi) * 256 + nt * 16 + n] = acc[vi] + bias;
}

// ---------------------------------------------------------------- launch
extern "C" void kernel_launch(void* const* d_in, const int* in_sizes, int n_in,
                              void* d_out, int out_size, void* d_ws, size_t ws_size,
                              hipStream_t stream) {
  const float* msa_act  = (const float*)d_in[0];
  const float* pair_act = (const float*)d_in[1];
  const float* msa_mask = (const float*)d_in[2];
  const float* qn_g     = (const float*)d_in[3];
  const float* qn_b     = (const float*)d_in[4];
  const float* pn_g     = (const float*)d_in[5];
  const float* pn_b     = (const float*)d_in[6];
  const float* w_pair   = (const float*)d_in[7];
  const float* wq       = (const float*)d_in[8];
  const float* wk       = (const float*)d_in[9];
  const float* wv       = (const float*)d_in[10];
  const float* wg       = (const float*)d_in[11];
  const float* bg       = (const float*)d_in[12];
  const float* wo       = (const float*)d_in[13];
  const float* bo       = (const float*)d_in[14];
  float* out = (float*)d_out;

  char* ws = (char*)d_ws;
  _Float16* wT_  = (_Float16*)(ws + 0);               // 1024*256*2      = 524288
  _Float16* woT_ = (_Float16*)(ws + 524288);          // 256*256*2       = 131072
  float*    z_   = (float*)   (ws + 655360);          // 8*384*384*4     = 4718592
  _Float16* m_   = (_Float16*)(ws + 5373952);         // 49152*256*2     = 25165824
  _Float16* q_   = (_Float16*)(ws + 30539776);        // 128*8*384*32*2  = 25165824
  _Float16* k_   = (_Float16*)(ws + 55705600);
  _Float16* vT_  = (_Float16*)(ws + 80871424);
  _Float16* gT_  = (_Float16*)(ws + 106037248);
  _Float16* og_  = (_Float16*)(ws + 131203072);       // 49152*256*2

  prep_weights_kernel<<<1280, 256, 0, stream>>>(wq, wk, wv, wg, wo, wT_, woT_);
  pair_bias_kernel<<<18432, 256, 0, stream>>>(pair_act, pn_g, pn_b, w_pair, z_);
  ln_msa_kernel<<<6144, 256, 0, stream>>>(msa_act, qn_g, qn_b, m_);
  qkvg_gemm_kernel<<<24576, 256, 0, stream>>>(m_, wT_, bg, q_, k_, vT_, gT_);
  attn_kernel<<<6144, 128, 0, stream>>>(q_, k_, vT_, gT_, z_, msa_mask, og_);
  out_gemm_kernel<<<6144, 256, 0, stream>>>(og_, woT_, bo, out);
}
